// Global_attention_Module_77129022702076
// MI455X (gfx1250) — matrix operand order repaired and checked by reading, not compiled, not run on hardware
//
#include <hip/hip_runtime.h>
#include <hip/hip_bf16.h>

// ---------------------------------------------------------------------------
// MI455X (gfx1250) implementation.
//  - All four GEMM stages use v_wmma_f32_16x16x32_bf16 (wave32, 16x16 tiles,
//    each wave owns a 32x32 output tile => acc = 2x2 v8f fragments).
//  - K/V are never materialized: diagKV and Ksum are reduced in the QKV GEMM
//    epilogue via cross-half-wave shuffle + atomics.
//  - Q1/K1 stored transposed (bf16) so the Q1^T*K1 GEMM reads row-major
//    fragments; softmax writes mid^T so the final GEMM's B-frags are
//    contiguous too.
//  - Workspace budget ~240 MB (Q fp32 buffer reused as y; xs bf16 reused as
//    result bf16).
// ---------------------------------------------------------------------------

typedef __bf16 v16bf __attribute__((ext_vector_type(16)));
typedef __bf16 v8bf  __attribute__((ext_vector_type(8)));
typedef float  v8f   __attribute__((ext_vector_type(8)));

static constexpr int DIM  = 512;
static constexpr int LTOK = 4096;
static constexpr int BATCH = 8;
static constexpr int MTOT = BATCH * LTOK;       // 32768
static constexpr int WELEM = DIM * DIM;          // 262144 elems per weight

__device__ __forceinline__ v8f wmma_bf16(v16bf a, v16bf b, v8f c) {
  return __builtin_amdgcn_wmma_f32_16x16x32_bf16(false, a, false, b,
                                                 (short)0, c, false, false);
}

// A-fragment (16x32 bf16, row-major source, k contiguous).
// lanes 0-15: row=lane, k in {0..7, 16..23}; lanes 16-31: row=lane-16,
// k in {8..15, 24..31}.
__device__ __forceinline__ v16bf load_a(const __bf16* __restrict__ A, int lda) {
  const int lane = threadIdx.x & 31;
  const int row  = lane & 15;
  const int kb   = (lane >> 4) << 3;             // 0 or 8
  const __bf16* p = A + (size_t)row * lda + kb;
  v8bf lo = *(const v8bf*)(p);
  v8bf hi = *(const v8bf*)(p + 16);
  v16bf r;
#pragma unroll
  for (int i = 0; i < 8; ++i) { r[i] = lo[i]; r[i + 8] = hi[i]; }
  return r;
}

// B-fragment (32x16 bf16) from Bt stored row-major as Bt[n][k] (k contiguous).
// lanes 0-15: col=lane, k in {0..7, 16..23}; lanes 16-31: col=lane-16,
// k in {8..15, 24..31}.
__device__ __forceinline__ v16bf load_bt(const __bf16* __restrict__ Bt, int ldb) {
  const int lane = threadIdx.x & 31;
  const int col  = lane & 15;
  const int kb   = (lane >> 4) << 3;             // 0 or 8
  const __bf16* p = Bt + (size_t)col * ldb + kb;
  v8bf lo = *(const v8bf*)(p);
  v8bf hi = *(const v8bf*)(p + 16);
  v16bf r;
#pragma unroll
  for (int i = 0; i < 8; ++i) { r[i] = lo[i]; r[i + 8] = hi[i]; }
  return r;
}

// ---------------------------------------------------------------------------
// 1) xs[b,l,c] = x[b,c,l] + pos[l,c]   (LDS tiled transpose, fp32 + bf16 out)
// ---------------------------------------------------------------------------
__global__ void build_xs_kernel(const float* __restrict__ x,
                                const float* __restrict__ pos,
                                float* __restrict__ xs_f,
                                __bf16* __restrict__ xs_b) {
  __shared__ float tile[32][33];
  const int b  = blockIdx.z;
  const int l0 = blockIdx.x * 32, c0 = blockIdx.y * 32;
  const int tx = threadIdx.x, ty = threadIdx.y;
  tile[ty][tx] = x[((size_t)(b * DIM + c0 + ty)) * LTOK + l0 + tx];
  __syncthreads();
  const int l = l0 + ty, c = c0 + tx;
  const float v = tile[tx][ty] + pos[(size_t)l * DIM + c];
  const size_t o = ((size_t)(b * LTOK + l)) * DIM + c;
  xs_f[o] = v;
  xs_b[o] = (__bf16)v;
}

// ---------------------------------------------------------------------------
// 2) Convert the five 512x512 weight matrices to bf16 (layout unchanged ==
//    exactly the Bt[n][k] layout the B-fragment loader wants, since the GEMM
//    computes xs @ W^T).
// ---------------------------------------------------------------------------
__global__ void conv_w_kernel(const float* __restrict__ w0, const float* __restrict__ w1,
                              const float* __restrict__ w2, const float* __restrict__ w3,
                              const float* __restrict__ w4, __bf16* __restrict__ dst) {
  const int idx = blockIdx.x * blockDim.x + threadIdx.x;   // 5*262144 total
  const int sel = idx >> 18;
  const int off = idx & (WELEM - 1);
  const float* s = (sel == 0) ? w0 : (sel == 1) ? w1 : (sel == 2) ? w2
                 : (sel == 3) ? w3 : w4;
  dst[idx] = (__bf16)s[off];
}

// ---------------------------------------------------------------------------
// 3) Fused Q/K/V GEMM. Q -> relu+store fp32. K,V -> only reduced into
//    diagKV[b,c] and Ksum[b,c] via shuffle + atomics (never stored).
// ---------------------------------------------------------------------------
__global__ __launch_bounds__(256)
void gemm_qkv_kernel(const __bf16* __restrict__ A, const __bf16* __restrict__ Wb,
                     const float* __restrict__ bq, const float* __restrict__ bk,
                     const float* __restrict__ bv, float* __restrict__ Qout,
                     float* __restrict__ diagKV, float* __restrict__ Ksum) {
  const int lane = threadIdx.x & 31;
  const int wt   = blockIdx.x * 8 + (threadIdx.x >> 5);
  const int tn   = wt & 15;                       // 16 n-tiles
  const int tm   = wt >> 4;                       // 1024 m-tiles
  const int m0 = tm * 32, n0 = tn * 32;
  const int b  = m0 >> 12;

  const __bf16* Wq = Wb;
  const __bf16* Wk = Wb + WELEM;
  const __bf16* Wv = Wb + 2 * WELEM;

  v8f aq[2][2] = {}; v8f ak[2][2] = {}; v8f av[2][2] = {};
  const __bf16* Am = A + (size_t)m0 * DIM;
  for (int k = 0; k < DIM; k += 32) {
    v16bf a0 = load_a(Am + k, DIM);
    v16bf a1 = load_a(Am + (size_t)16 * DIM + k, DIM);
    v16bf b0 = load_bt(Wq + (size_t)n0 * DIM + k, DIM);
    v16bf b1 = load_bt(Wq + (size_t)(n0 + 16) * DIM + k, DIM);
    aq[0][0] = wmma_bf16(a0, b0, aq[0][0]); aq[1][0] = wmma_bf16(a1, b0, aq[1][0]);
    aq[0][1] = wmma_bf16(a0, b1, aq[0][1]); aq[1][1] = wmma_bf16(a1, b1, aq[1][1]);
    b0 = load_bt(Wk + (size_t)n0 * DIM + k, DIM);
    b1 = load_bt(Wk + (size_t)(n0 + 16) * DIM + k, DIM);
    ak[0][0] = wmma_bf16(a0, b0, ak[0][0]); ak[1][0] = wmma_bf16(a1, b0, ak[1][0]);
    ak[0][1] = wmma_bf16(a0, b1, ak[0][1]); ak[1][1] = wmma_bf16(a1, b1, ak[1][1]);
    b0 = load_bt(Wv + (size_t)n0 * DIM + k, DIM);
    b1 = load_bt(Wv + (size_t)(n0 + 16) * DIM + k, DIM);
    av[0][0] = wmma_bf16(a0, b0, av[0][0]); av[1][0] = wmma_bf16(a1, b0, av[1][0]);
    av[0][1] = wmma_bf16(a0, b1, av[0][1]); av[1][1] = wmma_bf16(a1, b1, av[1][1]);
  }

  const int nc = lane & 15;
  const int rbase = (lane >> 4) * 8;
  float kv[2] = {0.f, 0.f}, ks[2] = {0.f, 0.f};
#pragma unroll
  for (int ni = 0; ni < 2; ++ni) {
    const int n = n0 + ni * 16 + nc;
    const float bqv = bq[n], bkv = bk[n], bvv = bv[n];
#pragma unroll
    for (int mi = 0; mi < 2; ++mi) {
#pragma unroll
      for (int i = 0; i < 8; ++i) {
        const int m = m0 + mi * 16 + rbase + i;
        float q = fmaxf(aq[mi][ni][i] + bqv, 0.f);
        Qout[(size_t)m * DIM + n] = q;
        float kkv = fmaxf(ak[mi][ni][i] + bkv, 0.f);
        float vvv = av[mi][ni][i] + bvv;
        kv[ni] += kkv * vvv;
        ks[ni] += kkv;
      }
    }
  }
#pragma unroll
  for (int ni = 0; ni < 2; ++ni) {
    float kvt = kv[ni] + __shfl_xor(kv[ni], 16, 32);
    float kst = ks[ni] + __shfl_xor(ks[ni], 16, 32);
    if (lane < 16) {
      atomicAdd(&diagKV[b * DIM + n0 + ni * 16 + nc], kvt);
      atomicAdd(&Ksum[b * DIM + n0 + ni * 16 + nc], kst);
    }
  }
}

// ---------------------------------------------------------------------------
// 4) Q1/K1 GEMM; epilogue stores bf16 TRANSPOSED: Q1t/K1t[b][c][l].
// ---------------------------------------------------------------------------
__global__ __launch_bounds__(256)
void gemm_q1k1_kernel(const __bf16* __restrict__ A, const __bf16* __restrict__ Wb,
                      const float* __restrict__ bq1, const float* __restrict__ bk1,
                      __bf16* __restrict__ Q1t, __bf16* __restrict__ K1t) {
  const int lane = threadIdx.x & 31;
  const int wt   = blockIdx.x * 8 + (threadIdx.x >> 5);
  const int tn   = wt & 15;
  const int tm   = wt >> 4;
  const int m0 = tm * 32, n0 = tn * 32;
  const int b  = m0 >> 12;
  const int ml0 = m0 & (LTOK - 1);

  const __bf16* Wq1 = Wb + 3 * WELEM;
  const __bf16* Wk1 = Wb + 4 * WELEM;

  v8f a1acc[2][2] = {}; v8f a2acc[2][2] = {};
  const __bf16* Am = A + (size_t)m0 * DIM;
  for (int k = 0; k < DIM; k += 32) {
    v16bf a0 = load_a(Am + k, DIM);
    v16bf a1 = load_a(Am + (size_t)16 * DIM + k, DIM);
    v16bf b0 = load_bt(Wq1 + (size_t)n0 * DIM + k, DIM);
    v16bf b1 = load_bt(Wq1 + (size_t)(n0 + 16) * DIM + k, DIM);
    a1acc[0][0] = wmma_bf16(a0, b0, a1acc[0][0]); a1acc[1][0] = wmma_bf16(a1, b0, a1acc[1][0]);
    a1acc[0][1] = wmma_bf16(a0, b1, a1acc[0][1]); a1acc[1][1] = wmma_bf16(a1, b1, a1acc[1][1]);
    b0 = load_bt(Wk1 + (size_t)n0 * DIM + k, DIM);
    b1 = load_bt(Wk1 + (size_t)(n0 + 16) * DIM + k, DIM);
    a2acc[0][0] = wmma_bf16(a0, b0, a2acc[0][0]); a2acc[1][0] = wmma_bf16(a1, b0, a2acc[1][0]);
    a2acc[0][1] = wmma_bf16(a0, b1, a2acc[0][1]); a2acc[1][1] = wmma_bf16(a1, b1, a2acc[1][1]);
  }

  const int nc = lane & 15;
  const int rbase = (lane >> 4) * 8;
#pragma unroll
  for (int ni = 0; ni < 2; ++ni) {
    const int n = n0 + ni * 16 + nc;
    const float bqv = bq1[n], bkv = bk1[n];
    __bf16* qrow = Q1t + ((size_t)(b * DIM + n)) * LTOK;
    __bf16* krow = K1t + ((size_t)(b * DIM + n)) * LTOK;
#pragma unroll
    for (int mi = 0; mi < 2; ++mi) {
#pragma unroll
      for (int i = 0; i < 8; ++i) {
        const int ml = ml0 + mi * 16 + rbase + i;
        qrow[ml] = (__bf16)(a1acc[mi][ni][i] + bqv);
        krow[ml] = (__bf16)(a2acc[mi][ni][i] + bkv);
      }
    }
  }
}

// ---------------------------------------------------------------------------
// 5) mid[b,c,d] = sum_l Q1[b,l,c]*K1[b,l,d]  (A=Q1t row-major, Bt=K1t)
// ---------------------------------------------------------------------------
__global__ __launch_bounds__(256)
void gemm_mid_kernel(const __bf16* __restrict__ Q1t, const __bf16* __restrict__ K1t,
                     float* __restrict__ mid) {
  const int lane = threadIdx.x & 31;
  const int wt   = blockIdx.x * 8 + (threadIdx.x >> 5);  // 0..2047
  const int b    = wt >> 8;
  const int r    = wt & 255;
  const int tm = r >> 4, tn = r & 15;
  const int m0 = tm * 32, n0 = tn * 32;
  const __bf16* Ab = Q1t + (size_t)b * DIM * LTOK + (size_t)m0 * LTOK;
  const __bf16* Bb = K1t + (size_t)b * DIM * LTOK + (size_t)n0 * LTOK;

  v8f acc[2][2] = {};
  for (int k = 0; k < LTOK; k += 32) {
    __builtin_prefetch(Ab + k + 256, 0, 1);
    __builtin_prefetch(Bb + k + 256, 0, 1);
    v16bf a0 = load_a(Ab + k, LTOK);
    v16bf a1 = load_a(Ab + (size_t)16 * LTOK + k, LTOK);
    v16bf b0 = load_bt(Bb + k, LTOK);
    v16bf b1 = load_bt(Bb + (size_t)16 * LTOK + k, LTOK);
    acc[0][0] = wmma_bf16(a0, b0, acc[0][0]); acc[1][0] = wmma_bf16(a1, b0, acc[1][0]);
    acc[0][1] = wmma_bf16(a0, b1, acc[0][1]); acc[1][1] = wmma_bf16(a1, b1, acc[1][1]);
  }

  const int nc = lane & 15;
  const int rbase = (lane >> 4) * 8;
  float* out = mid + (size_t)b * DIM * DIM;
#pragma unroll
  for (int ni = 0; ni < 2; ++ni)
#pragma unroll
    for (int mi = 0; mi < 2; ++mi)
#pragma unroll
      for (int i = 0; i < 8; ++i)
        out[(size_t)(m0 + mi * 16 + rbase + i) * DIM + n0 + ni * 16 + nc] = acc[mi][ni][i];
}

// ---------------------------------------------------------------------------
// 6) Row softmax of mid[b,c,:]; writes bf16 TRANSPOSED midT[b][d][c].
// ---------------------------------------------------------------------------
__global__ void softmax_mid_kernel(const float* __restrict__ mid,
                                   __bf16* __restrict__ midT) {
  const int row = blockIdx.x;                // b*512 + c
  const int b = row >> 9, c = row & 511;
  const float* p = mid + (size_t)row * DIM;
  const int tid = threadIdx.x;
  const int wave = tid >> 5, lane = tid & 31;
  __shared__ float red[8];

  float x0 = p[tid], x1 = p[tid + 256];
  float m = fmaxf(x0, x1);
#pragma unroll
  for (int off = 16; off; off >>= 1) m = fmaxf(m, __shfl_xor(m, off, 32));
  if (lane == 0) red[wave] = m;
  __syncthreads();
  if (wave == 0) {
    float t = red[lane & 7];
#pragma unroll
    for (int off = 4; off; off >>= 1) t = fmaxf(t, __shfl_xor(t, off, 32));
    if (lane == 0) red[0] = t;
  }
  __syncthreads();
  const float rmax = red[0];
  __syncthreads();

  float e0 = __expf(x0 - rmax), e1 = __expf(x1 - rmax);
  float s = e0 + e1;
#pragma unroll
  for (int off = 16; off; off >>= 1) s += __shfl_xor(s, off, 32);
  if (lane == 0) red[wave] = s;
  __syncthreads();
  if (wave == 0) {
    float t = red[lane & 7];
#pragma unroll
    for (int off = 4; off; off >>= 1) t += __shfl_xor(t, off, 32);
    if (lane == 0) red[0] = t;
  }
  __syncthreads();
  const float inv = 1.0f / red[0];

  midT[((size_t)(b * DIM + tid)) * DIM + c]       = (__bf16)(e0 * inv);
  midT[((size_t)(b * DIM + tid + 256)) * DIM + c] = (__bf16)(e1 * inv);
}

// ---------------------------------------------------------------------------
// 7) Z[b,l] = 1/(Q . (Ksum+eps));  resb[b,l,c] = bf16(Q*diagKV*Z)
//    One wave per token row (wave32 shuffle reduction).
// ---------------------------------------------------------------------------
__global__ void z_result_kernel(const float* __restrict__ Q,
                                const float* __restrict__ diagKV,
                                const float* __restrict__ Ksum,
                                __bf16* __restrict__ resb) {
  const int lane = threadIdx.x & 31;
  const int m = blockIdx.x * 8 + (threadIdx.x >> 5);
  const int b = m >> 12;
  const float* q  = Q + (size_t)m * DIM;
  const float* ks = Ksum + b * DIM;
  const float* dk = diagKV + b * DIM;
  float acc = 0.f;
#pragma unroll
  for (int j = 0; j < 16; ++j) {
    const int c = lane + j * 32;
    acc += q[c] * (ks[c] + 1e-6f);
  }
#pragma unroll
  for (int off = 16; off; off >>= 1) acc += __shfl_xor(acc, off, 32);
  const float z = 1.0f / acc;
  __bf16* r = resb + (size_t)m * DIM;
#pragma unroll
  for (int j = 0; j < 16; ++j) {
    const int c = lane + j * 32;
    r[c] = (__bf16)(q[c] * dk[c] * z);
  }
}

// ---------------------------------------------------------------------------
// 8) out = result @ mid;  y[b,l,c] = xs + gamma*out   (A=resb, Bt=midT)
// ---------------------------------------------------------------------------
__global__ __launch_bounds__(256)
void gemm_res_kernel(const __bf16* __restrict__ Rb, const __bf16* __restrict__ midT,
                     const float* __restrict__ xs_f, const float* __restrict__ gamma,
                     float* __restrict__ y) {
  const int lane = threadIdx.x & 31;
  const int wt   = blockIdx.x * 8 + (threadIdx.x >> 5);
  const int tn   = wt & 15;
  const int tm   = wt >> 4;
  const int m0 = tm * 32, n0 = tn * 32;
  const int b  = m0 >> 12;

  const __bf16* Am = Rb + (size_t)m0 * DIM;
  const __bf16* Bb = midT + (size_t)b * DIM * DIM + (size_t)n0 * DIM;

  v8f acc[2][2] = {};
  for (int k = 0; k < DIM; k += 32) {
    v16bf a0 = load_a(Am + k, DIM);
    v16bf a1 = load_a(Am + (size_t)16 * DIM + k, DIM);
    v16bf b0 = load_bt(Bb + k, DIM);
    v16bf b1 = load_bt(Bb + (size_t)16 * DIM + k, DIM);
    acc[0][0] = wmma_bf16(a0, b0, acc[0][0]); acc[1][0] = wmma_bf16(a1, b0, acc[1][0]);
    acc[0][1] = wmma_bf16(a0, b1, acc[0][1]); acc[1][1] = wmma_bf16(a1, b1, acc[1][1]);
  }

  const float g = gamma[0];
  const int nc = lane & 15;
  const int rbase = (lane >> 4) * 8;
#pragma unroll
  for (int ni = 0; ni < 2; ++ni)
#pragma unroll
    for (int mi = 0; mi < 2; ++mi)
#pragma unroll
      for (int i = 0; i < 8; ++i) {
        const size_t o = (size_t)(m0 + mi * 16 + rbase + i) * DIM + n0 + ni * 16 + nc;
        y[o] = xs_f[o] + g * acc[mi][ni][i];
      }
}

// ---------------------------------------------------------------------------
// 9) BatchNorm statistics: per-channel sum / sumsq over all 32768 tokens.
// ---------------------------------------------------------------------------
__global__ void bn_stats_kernel(const float* __restrict__ y,
                                float* __restrict__ csum, float* __restrict__ csq) {
  const int c = threadIdx.x;                 // 512 threads
  const int r0 = blockIdx.x * 64;
  float s = 0.f, q = 0.f;
  for (int r = 0; r < 64; ++r) {
    const float v = y[(size_t)(r0 + r) * DIM + c];
    s += v; q += v * v;
  }
  atomicAdd(&csum[c], s);
  atomicAdd(&csq[c], q);
}

__global__ void bn_finalize_kernel(const float* __restrict__ csum,
                                   const float* __restrict__ csq,
                                   const float* __restrict__ bnw,
                                   const float* __restrict__ bnb,
                                   float* __restrict__ scale,
                                   float* __restrict__ shift) {
  const int c = threadIdx.x;
  const float invN = 1.0f / (float)MTOT;
  const float mean = csum[c] * invN;
  const float var = csq[c] * invN - mean * mean;
  const float sc = rsqrtf(var + 1e-5f) * bnw[c];
  scale[c] = sc;
  shift[c] = bnb[c] - mean * sc;
}

// ---------------------------------------------------------------------------
// 10) out[b,c,h,w] = relu(y[b,l,c]*scale[c]+shift[c])  (LDS tiled transpose)
// ---------------------------------------------------------------------------
__global__ void bn_out_kernel(const float* __restrict__ y,
                              const float* __restrict__ scale,
                              const float* __restrict__ shift,
                              float* __restrict__ out) {
  __shared__ float tile[32][33];
  const int b  = blockIdx.z;
  const int l0 = blockIdx.x * 32, c0 = blockIdx.y * 32;
  const int tx = threadIdx.x, ty = threadIdx.y;
  tile[ty][tx] = y[((size_t)(b * LTOK + l0 + ty)) * DIM + c0 + tx];
  __syncthreads();
  const int c = c0 + ty;
  const float v = tile[tx][ty] * scale[c] + shift[c];
  out[((size_t)(b * DIM + c)) * LTOK + l0 + tx] = fmaxf(v, 0.f);
}

// ---------------------------------------------------------------------------
extern "C" void kernel_launch(void* const* d_in, const int* in_sizes, int n_in,
                              void* d_out, int out_size, void* d_ws, size_t ws_size,
                              hipStream_t stream) {
  (void)in_sizes; (void)n_in; (void)out_size; (void)ws_size;

  const float* x     = (const float*)d_in[0];
  const float* pos   = (const float*)d_in[1];
  const float* Wq    = (const float*)d_in[2];
  const float* bq    = (const float*)d_in[3];
  const float* Wk    = (const float*)d_in[4];
  const float* bk    = (const float*)d_in[5];
  const float* Wv    = (const float*)d_in[6];
  const float* bv    = (const float*)d_in[7];
  const float* Wq1   = (const float*)d_in[8];
  const float* bq1   = (const float*)d_in[9];
  const float* Wk1   = (const float*)d_in[10];
  const float* bk1   = (const float*)d_in[11];
  const float* gamma = (const float*)d_in[12];
  const float* bnw   = (const float*)d_in[13];
  const float* bnb   = (const float*)d_in[14];
  float* out = (float*)d_out;

  // Workspace layout (~240 MB)
  char* ws = (char*)d_ws;
  float*  xs_f  = (float*)(ws + 0);                 //  64 MB [32768,512] f32
  float*  Q_f   = (float*)(ws + (64ull << 20));     //  64 MB Q f32, reused as y
  __bf16* xs_b  = (__bf16*)(ws + (128ull << 20));   //  32 MB xs bf16, reused as result bf16
  __bf16* W_b   = (__bf16*)(ws + (160ull << 20));   // 2.5 MB 5x512x512 bf16
  __bf16* Q1t   = (__bf16*)(ws + (163ull << 20));   //  32 MB [8,512,4096] bf16
  __bf16* K1t   = (__bf16*)(ws + (195ull << 20));   //  32 MB
  float*  mid_f = (float*)(ws + (227ull << 20));    //   8 MB [8,512,512] f32
  __bf16* midT  = (__bf16*)(ws + (235ull << 20));   //   4 MB [8,512,512] bf16 (transposed)
  float*  stats = (float*)(ws + (239ull << 20));    //  40 KB accumulators
  float* diagKV = stats;             // 4096
  float* Ksum   = stats + 4096;      // 4096
  float* csum   = stats + 8192;      // 512
  float* csq    = stats + 8704;      // 512
  float* scale  = stats + 9216;      // 512
  float* shift  = stats + 9728;      // 512

  hipMemsetAsync(stats, 0, 40 * 1024, stream);

  // 1) xs build (transpose + pos add)
  build_xs_kernel<<<dim3(LTOK / 32, DIM / 32, BATCH), dim3(32, 32), 0, stream>>>(
      x, pos, xs_f, xs_b);

  // 2) weights -> bf16
  conv_w_kernel<<<(5 * WELEM) / 256, 256, 0, stream>>>(Wq, Wk, Wv, Wq1, Wk1, W_b);

  // 3) fused Q/K/V GEMM (+diagKV, Ksum)
  gemm_qkv_kernel<<<(MTOT / 32) * (DIM / 32) / 8, 256, 0, stream>>>(
      xs_b, W_b, bq, bk, bv, Q_f, diagKV, Ksum);

  // 4) Q1/K1 GEMM with transposed bf16 stores
  gemm_q1k1_kernel<<<(MTOT / 32) * (DIM / 32) / 8, 256, 0, stream>>>(
      xs_b, W_b, bq1, bk1, Q1t, K1t);

  // 5) mid = Q1^T @ K1 per batch
  gemm_mid_kernel<<<BATCH * (DIM / 32) * (DIM / 32) / 8, 256, 0, stream>>>(
      Q1t, K1t, mid_f);

  // 6) softmax rows -> midT (bf16)
  softmax_mid_kernel<<<BATCH * DIM, 256, 0, stream>>>(mid_f, midT);

  // 7) Z + result (bf16), overwrites xs_b
  z_result_kernel<<<MTOT / 8, 256, 0, stream>>>(Q_f, diagKV, Ksum, xs_b);

  // 8) result @ mid, residual: y = xs + gamma*res (y reuses Q_f)
  gemm_res_kernel<<<(MTOT / 32) * (DIM / 32) / 8, 256, 0, stream>>>(
      xs_b, midT, xs_f, gamma, Q_f);

  // 9) BN stats + finalize
  bn_stats_kernel<<<MTOT / 64, DIM, 0, stream>>>(Q_f, csum, csq);
  bn_finalize_kernel<<<1, DIM, 0, stream>>>(csum, csq, bnw, bnb, scale, shift);

  // 10) normalize + relu + NCHW transpose
  bn_out_kernel<<<dim3(LTOK / 32, DIM / 32, BATCH), dim3(32, 32), 0, stream>>>(
      Q_f, scale, shift, out);
}
